// sLSTMCell_9397388444340
// MI455X (gfx1250) — matrix operand order repaired and checked by reading, not compiled, not run on hardware
//
#include <hip/hip_runtime.h>
#include <hip/hip_bf16.h>

// ---------------------------------------------------------------------------
// sLSTM cell for MI455X (gfx1250): bf16 WMMA GEMM fed by the Tensor Data
// Mover (tensor_load_to_lds, double-buffered, s_wait_tensorcnt), then a
// fused gates + 2x LayerNorm pass.
// ---------------------------------------------------------------------------

typedef __bf16 bf16_t;
typedef bf16_t v16bf __attribute__((ext_vector_type(16)));
typedef float  v8f   __attribute__((ext_vector_type(8)));
typedef unsigned int u32x4 __attribute__((ext_vector_type(4)));
typedef int          i32x4 __attribute__((ext_vector_type(4)));
typedef int          i32x8 __attribute__((ext_vector_type(8)));

#define B_ROWS 8192
#define K_DIM  2048     // D + U
#define N_DIM  4096     // 4 gates * U
#define U_DIM  1024
#define MT     128
#define NT     128
#define KB     32
#define LDSP   40       // LDS row stride in bf16 elems (64B data + 16B TDM pad)

union Frag { v16bf v; uint4 q[2]; };

__device__ __forceinline__ unsigned short f2bf(float f) {
    unsigned u = __float_as_uint(f);
    unsigned r = u + 0x7FFFu + ((u >> 16) & 1u);   // round-to-nearest-even
    return (unsigned short)(r >> 16);
}

// --- TDM issue: 2D tile load, descriptor per CDNA5 ISA ---
__device__ __forceinline__ void tdm_load(unsigned long long ga, unsigned lds_off,
                                         i32x8 g1) {
    u32x4 g0;
    g0[0] = 1u;                                   // count=1 (valid), no gather
    g0[1] = lds_off;                              // lds_addr (bytes)
    g0[2] = (unsigned)ga;                         // global_addr[31:0]
    g0[3] = (unsigned)((ga >> 32) & 0x01FFFFFFu)  // global_addr[56:32]
          | 0x80000000u;                          // type=2 ("image") in [127:126]
    i32x4 z4 = {0, 0, 0, 0};
    i32x8 z8 = {0, 0, 0, 0, 0, 0, 0, 0};
    __builtin_amdgcn_tensor_load_to_lds(g0, g1, z4, z4, z8, 0);
}

// group1: data_size=2B | pad_enable | pad_interval=16 DWORDs | pad_amount=4 DWORDs,
// tensor_dim0 = 2048, tile = 32 x 128, tensor_dim0_stride = 2048
__device__ __forceinline__ i32x8 make_g1(int tensor_dim1) {
    i32x8 g1;
    g1[0] = (1 << 16) | (1 << 20) | (3 << 22) | (3 << 25);
    g1[1] = (int)((K_DIM & 0xFFFF) << 16);            // tensor_dim0 lo -> [63:48]
    g1[2] = (int)((tensor_dim1 & 0xFFFF) << 16);      // tensor_dim1 lo -> [95:80]
    g1[3] = (int)(KB << 16);                          // tile_dim0 = 32 -> [127:112]
    g1[4] = NT;                                       // tile_dim1 = 128
    g1[5] = K_DIM;                                    // tensor_dim0_stride
    g1[6] = 0;
    g1[7] = 0;
    return g1;
}

// ------------------- one-time fp32 -> bf16 conversions ----------------------
__global__ __launch_bounds__(256) void convertA(
    const float* __restrict__ X, const float* __restrict__ H,
    unsigned short* __restrict__ Abf)          // [8192][2048] bf16
{
    int gid = blockIdx.x * 256 + threadIdx.x;  // 8192*256 threads, 8 elems each
    int row = gid >> 8;
    int k0  = (gid & 255) * 8;
    const float* src = (k0 < 1024) ? (X + (size_t)row * 1024 + k0)
                                   : (H + (size_t)row * 1024 + (k0 - 1024));
    float4 f0 = *(const float4*)src;
    float4 f1 = *(const float4*)(src + 4);
    uint4 p;
    p.x = (unsigned)f2bf(f0.x) | ((unsigned)f2bf(f0.y) << 16);
    p.y = (unsigned)f2bf(f0.z) | ((unsigned)f2bf(f0.w) << 16);
    p.z = (unsigned)f2bf(f1.x) | ((unsigned)f2bf(f1.y) << 16);
    p.w = (unsigned)f2bf(f1.z) | ((unsigned)f2bf(f1.w) << 16);
    *(uint4*)&Abf[(size_t)row * K_DIM + k0] = p;
}

__global__ __launch_bounds__(256) void convertW(
    const float* __restrict__ Wi, const float* __restrict__ Wf,
    const float* __restrict__ Wo, const float* __restrict__ Wc,
    unsigned short* __restrict__ Wt)           // [4096][2048] bf16, [n][k]
{
    __shared__ float tile[32][33];
    const int k0 = blockIdx.x * 32;            // 64 tiles along K
    const int n0 = blockIdx.y * 32;            // 128 tiles along N
    const int gate = n0 >> 10;
    const int col0 = n0 & 1023;
    const float* Wg = (gate == 0) ? Wi : (gate == 1) ? Wf : (gate == 2) ? Wo : Wc;
    const int tx = threadIdx.x & 31;
    const int ty = threadIdx.x >> 5;
#pragma unroll
    for (int r = 0; r < 32; r += 8)            // coalesced read along n
        tile[ty + r][tx] = Wg[(size_t)(k0 + ty + r) * 1024 + col0 + tx];
    __syncthreads();
#pragma unroll
    for (int r = 0; r < 32; r += 8)            // Wt[n0+rr][k0+tx] = W[k0+tx][col0+rr]
        Wt[(size_t)(n0 + ty + r) * K_DIM + k0 + tx] = f2bf(tile[tx][ty + r]);
}

// ------------------------- GEMM: preact = A @ Wt^T --------------------------
__global__ __launch_bounds__(256) void slstm_gemm(
    const unsigned short* __restrict__ Abf,   // [8192][2048] bf16
    const unsigned short* __restrict__ Wt,    // [4096][2048] bf16
    float* __restrict__ preact)               // [8192][4096] fp32
{
    __shared__ unsigned short As[2][MT * LDSP];
    __shared__ unsigned short Bs[2][NT * LDSP];

    const int tid = threadIdx.x;
    const int m0  = blockIdx.y * MT;
    const int n0  = blockIdx.x * NT;

    const int lane  = tid & 31;
    const int wave  = tid >> 5;
    const int wm    = wave & 1;     // 2 waves along M (64 rows each)
    const int wn    = wave >> 1;    // 4 waves along N (32 cols each)
    const int mlane = lane & 15;
    const int aOff  = (lane >> 4) * 8;    // A K-offset (elements)
    const int bOff  = (lane >> 4) * 8;    // B K-offset (elements)

    const unsigned long long gaA =
        (unsigned long long)(uintptr_t)Abf + ((unsigned long long)m0 * K_DIM) * 2ull;
    const unsigned long long gaB =
        (unsigned long long)(uintptr_t)Wt + ((unsigned long long)n0 * K_DIM) * 2ull;

    const i32x8 g1A = make_g1(B_ROWS);
    const i32x8 g1B = make_g1(N_DIM);
    const unsigned ldsA0 = (unsigned)(uintptr_t)&As[0][0];
    const unsigned ldsA1 = (unsigned)(uintptr_t)&As[1][0];
    const unsigned ldsB0 = (unsigned)(uintptr_t)&Bs[0][0];
    const unsigned ldsB1 = (unsigned)(uintptr_t)&Bs[1][0];

    v8f acc[4][2];
#pragma unroll
    for (int i = 0; i < 4; ++i)
#pragma unroll
        for (int j = 0; j < 2; ++j)
#pragma unroll
            for (int r = 0; r < 8; ++r) acc[i][j][r] = 0.0f;

    const int NKI = K_DIM / KB;   // 64

    if (wave == 0) {              // prologue: tile 0 into buffer 0
        tdm_load(gaA, ldsA0, g1A);
        tdm_load(gaB, ldsB0, g1B);
    }

    for (int ki = 0; ki < NKI; ++ki) {
        if (wave == 0) {
            if (ki + 1 < NKI) {   // prefetch next tile into the other buffer
                unsigned long long koff = (unsigned long long)(ki + 1) * KB * 2ull;
                tdm_load(gaA + koff, ((ki + 1) & 1) ? ldsA1 : ldsA0, g1A);
                tdm_load(gaB + koff, ((ki + 1) & 1) ? ldsB1 : ldsB0, g1B);
                __builtin_amdgcn_s_wait_tensorcnt(2);  // current tile's 2 DMAs done
            } else {
                __builtin_amdgcn_s_wait_tensorcnt(0);
            }
        }
        __syncthreads();

        const unsigned short* Ac = As[ki & 1];
        const unsigned short* Bc = Bs[ki & 1];

        Frag a[4], bb[2];
#pragma unroll
        for (int i = 0; i < 4; ++i) {
            int r = (wm * 64 + i * 16 + mlane) * LDSP + aOff;
            a[i].q[0] = *(const uint4*)&Ac[r];        // K = aOff .. aOff+7
            a[i].q[1] = *(const uint4*)&Ac[r + 16];   // K = aOff+16 .. aOff+23
        }
#pragma unroll
        for (int j = 0; j < 2; ++j) {
            int r = (wn * 32 + j * 16 + mlane) * LDSP + bOff;
            bb[j].q[0] = *(const uint4*)&Bc[r];       // K = bOff .. bOff+7
            bb[j].q[1] = *(const uint4*)&Bc[r + 16];  // K = bOff+16 .. bOff+23
        }
#pragma unroll
        for (int i = 0; i < 4; ++i)
#pragma unroll
            for (int j = 0; j < 2; ++j)
                acc[i][j] = __builtin_amdgcn_wmma_f32_16x16x32_bf16(
                    false, a[i].v, false, bb[j].v, (short)0, acc[i][j], false, false);
        __syncthreads();
    }

    // ---- store C: lane = N column, VGPR r = M row (+8 for upper half-wave) --
    const int rbase = (lane >> 4) * 8;
#pragma unroll
    for (int i = 0; i < 4; ++i)
#pragma unroll
        for (int j = 0; j < 2; ++j) {
            size_t base = (size_t)(m0 + wm * 64 + i * 16 + rbase) * N_DIM
                        + (n0 + wn * 32 + j * 16 + mlane);
#pragma unroll
            for (int r = 0; r < 8; ++r)
                preact[base + (size_t)r * N_DIM] = acc[i][j][r];
        }
}

// ------------------------- gates + 2x LayerNorm per row ---------------------
__global__ __launch_bounds__(256) void slstm_pointwise(
    const float* __restrict__ preact, const float* __restrict__ c_prev,
    const float* __restrict__ bi, const float* __restrict__ ci,
    const float* __restrict__ bfv, const float* __restrict__ cf,
    const float* __restrict__ bo, const float* __restrict__ co,
    const float* __restrict__ bc,
    const float* __restrict__ gamma_c, const float* __restrict__ beta_c,
    const float* __restrict__ gamma_h, const float* __restrict__ beta_h,
    float* __restrict__ out)   // [h_new | c_new]
{
    __shared__ float2 red[256];
    const int m  = blockIdx.x;
    const int t  = threadIdx.x;
    const int u0 = t * 4;
    const size_t rowp = (size_t)m * N_DIM;

    float4 ip = *(const float4*)&preact[rowp + 0 * 1024 + u0];
    float4 fp = *(const float4*)&preact[rowp + 1 * 1024 + u0];
    float4 op = *(const float4*)&preact[rowp + 2 * 1024 + u0];
    float4 cp = *(const float4*)&preact[rowp + 3 * 1024 + u0];
    float4 cpv = *(const float4*)&c_prev[(size_t)m * U_DIM + u0];

    float z[4], og[4];
    float s = 0.f, s2 = 0.f;
#pragma unroll
    for (int e = 0; e < 4; ++e) {
        float xpre = ((const float*)&ip)[e] + bi[u0 + e];
        float fpre = ((const float*)&fp)[e] + bfv[u0 + e];
        float opre = ((const float*)&op)[e] + bo[u0 + e];
        float cc   = tanhf(((const float*)&cp)[e] + bc[u0 + e]);

        float ei = __expf(fminf(fmaxf(xpre, -10.f), 10.f));
        float ef = __expf(fminf(fmaxf(fpre, -10.f), 10.f));
        float eo = __expf(fminf(fmaxf(opre, -10.f), 10.f));
        float si = __expf(fminf(fmaxf(ci[u0 + e], -10.f), 10.f));
        float sf = __expf(fminf(fmaxf(cf[u0 + e], -10.f), 10.f));
        float so = __expf(fminf(fmaxf(co[u0 + e], -10.f), 10.f));

        float ig = ei / (1.f + ei + si);
        float fg = ef / (1.f + ef + sf);
        og[e]    = eo / (1.f + eo + so);

        z[e] = fg * ((const float*)&cpv)[e] + ig * cc;
        s += z[e]; s2 += z[e] * z[e];
    }

    // ---- LayerNorm #1 over 1024 elements ----
    red[t] = make_float2(s, s2);
    __syncthreads();
    for (int off = 128; off > 0; off >>= 1) {
        if (t < off) { red[t].x += red[t + off].x; red[t].y += red[t + off].y; }
        __syncthreads();
    }
    float mu  = red[0].x * (1.f / 1024.f);
    float var = red[0].y * (1.f / 1024.f) - mu * mu;
    float rs  = rsqrtf(var + 1e-3f);
    __syncthreads();

    float4 cnew, hnew;
    float tv[4];
    s = 0.f; s2 = 0.f;
#pragma unroll
    for (int e = 0; e < 4; ++e) {
        float cn = (z[e] - mu) * rs * gamma_c[u0 + e] + beta_c[u0 + e];
        ((float*)&cnew)[e] = cn;
        tv[e] = og[e] * tanhf(cn);
        s += tv[e]; s2 += tv[e] * tv[e];
    }

    // ---- LayerNorm #2 ----
    red[t] = make_float2(s, s2);
    __syncthreads();
    for (int off = 128; off > 0; off >>= 1) {
        if (t < off) { red[t].x += red[t + off].x; red[t].y += red[t + off].y; }
        __syncthreads();
    }
    mu  = red[0].x * (1.f / 1024.f);
    var = red[0].y * (1.f / 1024.f) - mu * mu;
    rs  = rsqrtf(var + 1e-3f);

#pragma unroll
    for (int e = 0; e < 4; ++e)
        ((float*)&hnew)[e] = (tv[e] - mu) * rs * gamma_h[u0 + e] + beta_h[u0 + e];

    *(float4*)&out[(size_t)m * U_DIM + u0] = hnew;                            // h_new
    *(float4*)&out[(size_t)B_ROWS * U_DIM + (size_t)m * U_DIM + u0] = cnew;   // c_new
}

// ---------------------------------------------------------------------------
extern "C" void kernel_launch(void* const* d_in, const int* in_sizes, int n_in,
                              void* d_out, int out_size, void* d_ws, size_t ws_size,
                              hipStream_t stream) {
    const float* X  = (const float*)d_in[0];
    const float* H  = (const float*)d_in[1];
    const float* Cp = (const float*)d_in[2];
    const float* Wi = (const float*)d_in[3];
    const float* bi = (const float*)d_in[4];
    const float* ci = (const float*)d_in[5];
    const float* Wf = (const float*)d_in[6];
    const float* bfv= (const float*)d_in[7];
    const float* cf = (const float*)d_in[8];
    const float* Wo = (const float*)d_in[9];
    const float* bo = (const float*)d_in[10];
    const float* co = (const float*)d_in[11];
    const float* Wc = (const float*)d_in[12];
    const float* bc = (const float*)d_in[13];
    const float* gc = (const float*)d_in[14];
    const float* bcn= (const float*)d_in[15];
    const float* gh = (const float*)d_in[16];
    const float* bh = (const float*)d_in[17];

    float* out = (float*)d_out;

    // ws layout: preact fp32 (128MB) | Abf bf16 (32MB) | Wt bf16 (16MB)
    float* preact = (float*)d_ws;
    unsigned short* Abf = (unsigned short*)((char*)d_ws +
                          (size_t)B_ROWS * N_DIM * sizeof(float));
    unsigned short* Wt  = Abf + (size_t)B_ROWS * K_DIM;

    convertA<<<8192, 256, 0, stream>>>(X, H, Abf);

    dim3 wGrid(K_DIM / 32, N_DIM / 32);   // 64 x 128
    convertW<<<wGrid, 256, 0, stream>>>(Wi, Wf, Wo, Wc, Wt);

    dim3 gGrid(N_DIM / NT, B_ROWS / MT);  // 32 x 64
    slstm_gemm<<<gGrid, 256, 0, stream>>>(Abf, Wt, preact);

    slstm_pointwise<<<B_ROWS, 256, 0, stream>>>(preact, Cp, bi, ci, bfv, cf,
                                                bo, co, bc, gc, bcn, gh, bh, out);
}
